// GINE_53635551592804
// MI455X (gfx1250) — matrix operand order repaired and checked by reading, not compiled, not run on hardware
//
#include <hip/hip_runtime.h>
#include <math.h>

typedef __bf16 bf16_t;
typedef __attribute__((ext_vector_type(16))) __bf16 v16bf;
typedef __attribute__((ext_vector_type(8)))  __bf16 v8bf;
typedef __attribute__((ext_vector_type(16))) float  v16f;
typedef __attribute__((ext_vector_type(8)))  float  v8f;
typedef __attribute__((ext_vector_type(4)))  float  v4f;

#define DD 128   // node feature dim
#define EDD 64   // edge feature dim
#define NLAYER 3

// float -> bf16 via native fptrunc (RNE); lowers to v_cvt_pk_bf16_f32
__device__ __forceinline__ bf16_t f2bf(float f) { return (bf16_t)f; }

// HW fp32 atomic add, GVS addressing: saddr base (SGPR pair) + 32-bit byte offset
// in a single VGPR -> no per-element 64-bit address math.
__device__ __forceinline__ void gatomic_add(float* base, int byteOff, float v) {
  asm volatile("global_atomic_add_f32 %0, %1, %2"
               :: "v"(byteOff), "v"(v), "s"(base) : "memory");
}

// -----------------------------------------------------------------------------
// Swizzle fp32 weights [L][K][128] into bf16 WMMA B-operand layout:
// flat = layer*K*128 + ((ntile*(K/32) + kk)*32 + lane)*16 + e
// where lane e element = B[kk*32 + ((lane<16)?0:8) + (e&7) + ((e>>3)<<4)][ntile*16 + (lane&15)]
// so each wave's B operand is one contiguous, coalesced 1 KB block.
// -----------------------------------------------------------------------------
__global__ __launch_bounds__(256) void swizzle_b_kernel(
    const float* __restrict__ src, bf16_t* __restrict__ dst,
    int perLog2 /* log2(K*128) */, int nkkLog2 /* log2(K/32) */, int total) {
  int idx = blockIdx.x * 256 + threadIdx.x;
  if (idx >= total) return;
  int layer = idx >> perLog2;
  int rem   = idx & ((1 << perLog2) - 1);
  int k = rem >> 7;          // row (K dim)
  int n = rem & 127;         // col (N dim)
  int nt = n >> 4;
  int kk = k >> 5;
  int kr = k & 31;
  int lane = (((kr >> 3) & 1) << 4) + (n & 15);
  int e = (kr & 7) + ((kr >> 4) << 3);
  int pos = (layer << perLog2) + (((((nt << nkkLog2) + kk) * 32 + lane) << 4) + e);
  dst[pos] = f2bf(src[idx]);
}

// -----------------------------------------------------------------------------
// Edge kernel: per wave, one 16-edge tile.
//   e   = edge_attr[16xED] @ We[EDx128] + be            (WMMA bf16, fp32 acc)
//   msg = relu(h[src] + e); agg[dst] += msg             (L2-resident gather/atomic)
// edge_attr is streamed exactly once -> non-temporal loads so the 192 MB L2
// keeps the node tables (hin 25.6 MB, agg 25.6 MB) resident for the gather /
// atomic traffic. The 16 src/dst indices are wave-uniform -> scalar SMEM loads
// + uniform SALU shifts; only the final half-select is per-lane.
// -----------------------------------------------------------------------------
__global__ __launch_bounds__(256) void gine_edge_kernel(
    const float* __restrict__ edge_attr,   // [E, 64] fp32
    const int*   __restrict__ srcIdx,      // [E]
    const int*   __restrict__ dstIdx,      // [E]
    const float* __restrict__ hin,         // [N, 128] fp32
    const bf16_t* __restrict__ WeB,        // swizzled [8][2][32][16]
    const float* __restrict__ be,          // [128]
    float* __restrict__ agg,               // [N, 128]
    int E) {
  const int lane = threadIdx.x & 31;
  const int wave = threadIdx.x >> 5;
  const int e0   = (blockIdx.x * 8 + wave) * 16;
  if (e0 >= E) return;                     // uniform per wave
  const int l15  = lane & 15;
  const int kb   = (lane < 16) ? 0 : 8;    // A-matrix K offset
  const bool lo16 = (lane < 16);

  // 16 src (element) / dst (byte) row offsets: wave-uniform scalar loads,
  // shifts done on the uniform (SALU) side; per-lane work = one select.
  const int e0u = __builtin_amdgcn_readfirstlane(e0);
  int soff[8], doff[8];
#pragma unroll
  for (int j = 0; j < 8; ++j) {
    int s_lo = srcIdx[e0u + j]     << 7;   // * 128 elements (uniform -> s_lshl)
    int s_hi = srcIdx[e0u + j + 8] << 7;
    int d_lo = dstIdx[e0u + j]     << 9;   // * 512 bytes   (uniform -> s_lshl)
    int d_hi = dstIdx[e0u + j + 8] << 9;
    soff[j] = lo16 ? s_lo : s_hi;
    doff[j] = lo16 ? d_lo : d_hi;
  }

  // Build A operands (16 edges x 64 feats, fp32 -> bf16, ISA 16-bit A layout)
  const float* ea = edge_attr + (size_t)(e0 + l15) * EDD;
  v16bf a[2];
#pragma unroll
  for (int kk = 0; kk < 2; ++kk) {
    const int base = kk * 32 + kb;
    v4f f0 = __builtin_nontemporal_load((const v4f*)(ea + base));
    v4f f1 = __builtin_nontemporal_load((const v4f*)(ea + base + 4));
    v4f g0 = __builtin_nontemporal_load((const v4f*)(ea + base + 16));
    v4f g1 = __builtin_nontemporal_load((const v4f*)(ea + base + 20));
    v16f t;
#pragma unroll
    for (int i = 0; i < 4; ++i) {
      t[i] = f0[i]; t[4 + i] = f1[i]; t[8 + i] = g0[i]; t[12 + i] = g1[i];
    }
    a[kk] = __builtin_convertvector(t, v16bf);   // packed v_cvt_pk_bf16_f32
  }

  // Column tiles in pairs: two independent accumulators -> minimal hazard NOPs
#pragma unroll
  for (int nt = 0; nt < 8; nt += 2) {
    const int col0 = nt * 16 + l15;
    const int col1 = col0 + 16;
    const float bias0 = be[col0];
    const float bias1 = be[col1];
    v8f c0 = {bias0, bias0, bias0, bias0, bias0, bias0, bias0, bias0};
    v8f c1 = {bias1, bias1, bias1, bias1, bias1, bias1, bias1, bias1};
#pragma unroll
    for (int kk = 0; kk < 2; ++kk) {
      v16bf b0 = *(const v16bf*)(WeB + ((((nt + 0) * 2 + kk) * 32 + lane) << 4));
      v16bf b1 = *(const v16bf*)(WeB + ((((nt + 1) * 2 + kk) * 32 + lane) << 4));
      c0 = __builtin_amdgcn_wmma_f32_16x16x32_bf16(false, a[kk], false, b0,
                                                   (short)0, c0, false, false);
      c1 = __builtin_amdgcn_wmma_f32_16x16x32_bf16(false, a[kk], false, b1,
                                                   (short)0, c1, false, false);
    }
    // Fused epilogue: gather h[src], relu, scatter-add to agg[dst]
    const int cb0 = col0 << 2;   // byte offsets within a row
    const int cb1 = col1 << 2;
#pragma unroll
    for (int j = 0; j < 8; ++j) {
      const int g = soff[j] + col0;        // col1 = g + 16 -> offset:64 form
      float m0 = c0[j] + hin[g];
      float m1 = c1[j] + hin[g + 16];
      gatomic_add(agg, doff[j] + cb0, fmaxf(m0, 0.0f));
      gatomic_add(agg, doff[j] + cb1, fmaxf(m1, 0.0f));
    }
  }
}

// -----------------------------------------------------------------------------
// Node kernel: per wave, one 16-row tile; fused 2-layer MLP on (agg + h).
//   ht   = relu((agg+h) @ W1 + b1)    -> staged bf16 in wave-private LDS
//   hout = ht @ W2 + b2  (+ relu between layers; NT store on final layer)
// -----------------------------------------------------------------------------
__global__ __launch_bounds__(256) void gine_node_kernel(
    const float* __restrict__ agg,
    const float* __restrict__ hin,
    const bf16_t* __restrict__ W1B,   // swizzled [8][4][32][16]
    const bf16_t* __restrict__ W2B,
    const float* __restrict__ b1,
    const float* __restrict__ b2,
    float* __restrict__ hout,
    int N, int relu_out) {
  __shared__ bf16_t ht[8][16][DD];    // 32 KB, wave-private slices
  const int lane = threadIdx.x & 31;
  const int wave = threadIdx.x >> 5;
  const int row0 = (blockIdx.x * 8 + wave) * 16;
  if (row0 >= N) return;              // uniform per wave (N % 16 == 0)
  const int l15  = lane & 15;
  const int half = (lane & 16) >> 1;
  const int kb   = (lane < 16) ? 0 : 8;

  // A = bf16(agg + h), in WMMA A layout, 4 K-steps of 32
  const float* ga = agg + (size_t)(row0 + l15) * DD;
  const float* gh = hin + (size_t)(row0 + l15) * DD;
  v16bf a[4];
#pragma unroll
  for (int kk = 0; kk < 4; ++kk) {
    const int base = kk * 32 + kb;
    v4f f0 = *(const v4f*)(ga + base),      h0 = *(const v4f*)(gh + base);
    v4f f1 = *(const v4f*)(ga + base + 4),  h1 = *(const v4f*)(gh + base + 4);
    v4f g0 = *(const v4f*)(ga + base + 16), i0 = *(const v4f*)(gh + base + 16);
    v4f g1 = *(const v4f*)(ga + base + 20), i1 = *(const v4f*)(gh + base + 20);
    v16f t;
#pragma unroll
    for (int i = 0; i < 4; ++i) {
      t[i]      = f0[i] + h0[i];
      t[4 + i]  = f1[i] + h1[i];
      t[8 + i]  = g0[i] + i0[i];
      t[12 + i] = g1[i] + i1[i];
    }
    a[kk] = __builtin_convertvector(t, v16bf);   // packed v_cvt_pk_bf16_f32
  }

  // GEMM1 + bias + relu -> LDS staging (same-wave LDS is in-order: no barrier)
#pragma unroll
  for (int nt = 0; nt < 8; nt += 2) {
    const int col0 = nt * 16 + l15;
    const int col1 = col0 + 16;
    const float bias0 = b1[col0];
    const float bias1 = b1[col1];
    v8f c0 = {bias0, bias0, bias0, bias0, bias0, bias0, bias0, bias0};
    v8f c1 = {bias1, bias1, bias1, bias1, bias1, bias1, bias1, bias1};
#pragma unroll
    for (int kk = 0; kk < 4; ++kk) {
      v16bf b0  = *(const v16bf*)(W1B + ((((nt + 0) * 4 + kk) * 32 + lane) << 4));
      v16bf b1v = *(const v16bf*)(W1B + ((((nt + 1) * 4 + kk) * 32 + lane) << 4));
      c0 = __builtin_amdgcn_wmma_f32_16x16x32_bf16(false, a[kk], false, b0,
                                                   (short)0, c0, false, false);
      c1 = __builtin_amdgcn_wmma_f32_16x16x32_bf16(false, a[kk], false, b1v,
                                                   (short)0, c1, false, false);
    }
#pragma unroll
    for (int j = 0; j < 8; ++j) {
      ht[wave][j + half][col0] = f2bf(fmaxf(c0[j], 0.0f));
      ht[wave][j + half][col1] = f2bf(fmaxf(c1[j], 0.0f));
    }
  }

  // Reload ht as A operand for GEMM2
  v16bf a2[4];
#pragma unroll
  for (int kk = 0; kk < 4; ++kk) {
    const int base = kk * 32 + kb;
    v8bf lo = *(const v8bf*)(&ht[wave][l15][base]);
    v8bf hi = *(const v8bf*)(&ht[wave][l15][base + 16]);
#pragma unroll
    for (int i = 0; i < 8; ++i) { a2[kk][i] = lo[i]; a2[kk][8 + i] = hi[i]; }
  }

#pragma unroll
  for (int nt = 0; nt < 8; nt += 2) {
    const int col0 = nt * 16 + l15;
    const int col1 = col0 + 16;
    const float bias0 = b2[col0];
    const float bias1 = b2[col1];
    v8f c0 = {bias0, bias0, bias0, bias0, bias0, bias0, bias0, bias0};
    v8f c1 = {bias1, bias1, bias1, bias1, bias1, bias1, bias1, bias1};
#pragma unroll
    for (int kk = 0; kk < 4; ++kk) {
      v16bf b0  = *(const v16bf*)(W2B + ((((nt + 0) * 4 + kk) * 32 + lane) << 4));
      v16bf b1v = *(const v16bf*)(W2B + ((((nt + 1) * 4 + kk) * 32 + lane) << 4));
      c0 = __builtin_amdgcn_wmma_f32_16x16x32_bf16(false, a2[kk], false, b0,
                                                   (short)0, c0, false, false);
      c1 = __builtin_amdgcn_wmma_f32_16x16x32_bf16(false, a2[kk], false, b1v,
                                                   (short)0, c1, false, false);
    }
    if (relu_out) {
      // Intermediate layer: relu, regular store (re-read by next layer -> RT)
#pragma unroll
      for (int j = 0; j < 8; ++j) {
        const size_t r = (size_t)(row0 + j + half) * DD;
        hout[r + col0] = fmaxf(c0[j], 0.0f);
        hout[r + col1] = fmaxf(c1[j], 0.0f);
      }
    } else {
      // Final layer: d_out is never re-read -> non-temporal stores
#pragma unroll
      for (int j = 0; j < 8; ++j) {
        const size_t r = (size_t)(row0 + j + half) * DD;
        __builtin_nontemporal_store(c0[j], hout + r + col0);
        __builtin_nontemporal_store(c1[j], hout + r + col1);
      }
    }
  }
}

// -----------------------------------------------------------------------------
extern "C" void kernel_launch(void* const* d_in, const int* in_sizes, int n_in,
                              void* d_out, int out_size, void* d_ws, size_t ws_size,
                              hipStream_t stream) {
  const float* x         = (const float*)d_in[0];   // [N,128]
  const int*   edge_index= (const int*)  d_in[1];   // [2,E]
  const float* edge_attr = (const float*)d_in[2];   // [E,64]
  const float* We        = (const float*)d_in[3];   // [L,64,128]
  const float* be        = (const float*)d_in[4];   // [L,128]
  const float* W1        = (const float*)d_in[5];   // [L,128,128]
  const float* b1        = (const float*)d_in[6];   // [L,128]
  const float* W2        = (const float*)d_in[7];   // [L,128,128]
  const float* b2        = (const float*)d_in[8];   // [L,128]

  const int N = in_sizes[0] / DD;
  const int E = in_sizes[1] / 2;
  const int* srcIdx = edge_index;
  const int* dstIdx = edge_index + E;

  // Workspace carve (all chunks 256B-aligned by construction)
  char* p = (char*)d_ws;
  const size_t nodeBytes = (size_t)N * DD * sizeof(float);
  float* hA  = (float*)p; p += nodeBytes;
  float* hB  = (float*)p; p += nodeBytes;
  float* agg = (float*)p; p += nodeBytes;
  bf16_t* WeB = (bf16_t*)p; p += (size_t)NLAYER * EDD * DD * sizeof(bf16_t);
  bf16_t* W1B = (bf16_t*)p; p += (size_t)NLAYER * DD * DD * sizeof(bf16_t);
  bf16_t* W2B = (bf16_t*)p; p += (size_t)NLAYER * DD * DD * sizeof(bf16_t);

  // h0 = x (do not mutate inputs)
  hipMemcpyAsync(hA, x, nodeBytes, hipMemcpyDeviceToDevice, stream);

  // Convert + swizzle weights to bf16 B-operand layout (cheap, once per call)
  {
    int totWe = NLAYER * EDD * DD;   // 24576,  per = 8192  -> log2 = 13, nkk=2 -> 1
    int totW  = NLAYER * DD * DD;    // 49152,  per = 16384 -> log2 = 14, nkk=4 -> 2
    swizzle_b_kernel<<<(totWe + 255) / 256, 256, 0, stream>>>(We, WeB, 13, 1, totWe);
    swizzle_b_kernel<<<(totW  + 255) / 256, 256, 0, stream>>>(W1, W1B, 14, 2, totW);
    swizzle_b_kernel<<<(totW  + 255) / 256, 256, 0, stream>>>(W2, W2B, 14, 2, totW);
  }

  const int edgeBlocks = (E + 127) / 128;              // 8 waves x 16 edges
  const int nodeBlocks = ((N + 15) / 16 + 7) / 8;      // 8 waves x 16 rows

  float* hin = hA;
  float* hother = hB;
  for (int l = 0; l < NLAYER; ++l) {
    hipMemsetAsync(agg, 0, nodeBytes, stream);
    gine_edge_kernel<<<edgeBlocks, 256, 0, stream>>>(
        edge_attr, srcIdx, dstIdx, hin,
        WeB + (size_t)l * EDD * DD, be + l * DD, agg, E);
    float* out = (l == NLAYER - 1) ? (float*)d_out : hother;
    gine_node_kernel<<<nodeBlocks, 256, 0, stream>>>(
        agg, hin,
        W1B + (size_t)l * DD * DD, W2B + (size_t)l * DD * DD,
        b1 + l * DD, b2 + l * DD, out, N, (l < NLAYER - 1) ? 1 : 0);
    hother = hin;
    hin = out;
  }
}
